// HierarchicalMemoryBank_850403525344
// MI455X (gfx1250) — matrix operand order repaired and checked by reading, not compiled, not run on hardware
//
#include <hip/hip_runtime.h>
#include <hip/hip_bf16.h>
#include <cstdint>
#include <cstddef>

// ---------------------------------------------------------------------------
// HierarchicalMemoryBank for MI455X (gfx1250, wave32, WMMA)
//   scores = q @ K^T / 32 + salience -> bf16 WMMA (f32 accum), 32x64 wave tile
//   per-row top-4 per level          -> LDS tile spill + full-wave lane scan
//     (two waves split the slot range; candidate lists merged in pass 3)
//   softmax(top4) @ gathered V rows  -> f32, L2-resident gathers
// ---------------------------------------------------------------------------

typedef __attribute__((ext_vector_type(16))) __bf16 v16bf;
typedef __attribute__((ext_vector_type(4)))  __bf16 v4bf;
typedef __attribute__((ext_vector_type(8)))  float  v8f;

#define D_MODEL     1024
#define NROWS       32768        // B*T = 8*4096
#define ROWS_PER_WG 128
#define KTOT        3584         // 2048 + 1024 + 512
#define QSTRIDE     1048         // halves; 12-dword row skew -> conflict-free b128 DS loads
#define QLDS_BYTES  (ROWS_PER_WG * QSTRIDE * 2)          // 268288
#define SCR_BYTES   (8 * 32 * 32 * 4)                    // 8 waves * 32x32 f32 chunk
#define SMEM_BYTES  (QLDS_BYTES + SCR_BYTES)             // 301056 < 320KB WGP LDS

union ABFrag { v16bf v; uint4 u[2]; };

#define WMMA_BF16(A, B, C) \
    __builtin_amdgcn_wmma_f32_16x16x32_bf16(false, (A), false, (B), (short)0, (C), false, false)

// ---------------- pass 1: f32 -> bf16 convert (K banks) ---------------------
__global__ void hmb_cvt_bf16(const float* __restrict__ src,
                             __bf16* __restrict__ dst, int n4) {
    int i = blockIdx.x * blockDim.x + threadIdx.x;
    if (i < n4) {
        float4 f = ((const float4*)src)[i];
        v4bf h = { (__bf16)f.x, (__bf16)f.y, (__bf16)f.z, (__bf16)f.w };
        *(v4bf*)(dst + (size_t)i * 4) = h;
    }
}

// ---------------- pass 2: WMMA scores + running top-4 -----------------------
__global__ __launch_bounds__(256)
void hmb_scores_topk(const float*  __restrict__ q,     // [NROWS][D] f32
                     const __bf16* __restrict__ Kb,    // [KTOT][D] bf16 (concat levels)
                     const float*  __restrict__ s0,
                     const float*  __restrict__ s1,
                     const float*  __restrict__ s2,
                     float* __restrict__ topS,         // [3][NROWS][2][4]
                     int*   __restrict__ topI)         // [3][NROWS][2][4]
{
    extern __shared__ char smem[];
    __bf16* qlds = (__bf16*)smem;
    float*  scr  = (float*)(smem + QLDS_BYTES);

    const int tid     = threadIdx.x;
    const int wave    = tid >> 5;
    const int lane    = tid & 31;
    const int rowBase = blockIdx.x * ROWS_PER_WG;

    // ---- stage 128 q rows into LDS as bf16 (vectorized f32x4 -> bf16x4) ----
    for (int i = tid; i < ROWS_PER_WG * (D_MODEL / 4); i += 256) {
        int r  = i / (D_MODEL / 4);
        int d4 = i % (D_MODEL / 4);
        float4 f = ((const float4*)(q + (size_t)(rowBase + r) * D_MODEL))[d4];
        v4bf h = { (__bf16)f.x, (__bf16)f.y, (__bf16)f.z, (__bf16)f.w };
        *(v4bf*)&qlds[(size_t)r * QSTRIDE + d4 * 4] = h;
    }
    __syncthreads();

    const int wquad   = wave & 3;         // which 32-row group of the block
    const int wgrp    = wave >> 2;        // which slot half (0/1)
    const int waveRow = wquad * 32;
    const int rloc    = lane & 15;        // A: row in frag / C: column / B: slot col
    const int aoff    = (lane >> 4) * 16; // A-frag k sub-offset (halves)
    const int boff    = (lane >> 4) * 16; // B-frag k sub-offset (halves)
    float* scrw = scr + wave * (32 * 32); // wave-private 32x32 f32 spill chunk

    const __bf16* arow0 = qlds + (size_t)(waveRow + rloc) * QSTRIDE + aoff;
    const __bf16* arow1 = arow0 + (size_t)16 * QSTRIDE;

    #pragma unroll
    for (int level = 0; level < 3; ++level) {
        const int    slotOff = (level == 0) ? 0    : (level == 1) ? 2048 : 3072;
        const int    nslots  = (level == 0) ? 2048 : (level == 1) ? 1024 : 512;
        const float* sal     = (level == 0) ? s0   : (level == 1) ? s1   : s2;

        const int halfSlots = nslots >> 1;        // per slot-group range
        const int idx0      = wgrp * halfSlots;   // index base within level
        const int slot0     = slotOff + idx0;     // row base within Kb

        float ts0 = -3.4e38f, ts1 = -3.4e38f, ts2 = -3.4e38f, ts3 = -3.4e38f;
        int   ti0 = 0, ti1 = 0, ti2 = 0, ti3 = 0;

        for (int sb = 0; sb < halfSlots; sb += 64) {   // 64-slot pass
            // B row pointers: 4 adjacent 16-slot tiles, per-lane one K row
            const __bf16* brow0 =
                Kb + (size_t)(slot0 + sb + rloc) * D_MODEL + boff;
            const __bf16* brow1 = brow0 + (size_t)16 * D_MODEL;
            const __bf16* brow2 = brow0 + (size_t)32 * D_MODEL;
            const __bf16* brow3 = brow0 + (size_t)48 * D_MODEL;

            v8f cA0 = {0.f,0.f,0.f,0.f,0.f,0.f,0.f,0.f};   // rows 0-15
            v8f cA1 = cA0, cA2 = cA0, cA3 = cA0;
            v8f cB0 = cA0, cB1 = cA0, cB2 = cA0, cB3 = cA0; // rows 16-31

            for (int kk = 0; kk < D_MODEL; kk += 32) {
                ABFrag a0, a1, b0, b1, b2, b3;
                a0.u[0] = *(const uint4*)(arow0 + kk);     // A reused by 4 WMMAs
                a0.u[1] = *(const uint4*)(arow0 + kk + 8);
                a1.u[0] = *(const uint4*)(arow1 + kk);
                a1.u[1] = *(const uint4*)(arow1 + kk + 8);
                b0.u[0] = *(const uint4*)(brow0 + kk);     // B reused by 2 WMMAs
                b0.u[1] = *(const uint4*)(brow0 + kk + 8);
                b1.u[0] = *(const uint4*)(brow1 + kk);
                b1.u[1] = *(const uint4*)(brow1 + kk + 8);
                b2.u[0] = *(const uint4*)(brow2 + kk);
                b2.u[1] = *(const uint4*)(brow2 + kk + 8);
                b3.u[0] = *(const uint4*)(brow3 + kk);
                b3.u[1] = *(const uint4*)(brow3 + kk + 8);
                cA0 = WMMA_BF16(a0.v, b0.v, cA0);
                cB0 = WMMA_BF16(a1.v, b0.v, cB0);
                cA1 = WMMA_BF16(a0.v, b1.v, cA1);
                cB1 = WMMA_BF16(a1.v, b1.v, cB1);
                cA2 = WMMA_BF16(a0.v, b2.v, cA2);
                cB2 = WMMA_BF16(a1.v, b2.v, cB2);
                cA3 = WMMA_BF16(a0.v, b3.v, cA3);
                cB3 = WMMA_BF16(a1.v, b3.v, cB3);
            }

            // spill + scan in two 32-row x 32-col chunks (keeps scratch small)
            const int m = (lane >> 4) << 3;               // C layout row base
            #pragma unroll
            for (int ch = 0; ch < 2; ++ch) {
                const v8f* pA0 = (ch == 0) ? &cA0 : &cA2;
                const v8f* pB0 = (ch == 0) ? &cB0 : &cB2;
                const v8f* pA1 = (ch == 0) ? &cA1 : &cA3;
                const v8f* pB1 = (ch == 0) ? &cB1 : &cB3;
                float sv0 = sal[idx0 + sb + ch * 32 + rloc];
                float sv1 = sal[idx0 + sb + ch * 32 + 16 + rloc];
                #pragma unroll
                for (int j = 0; j < 8; ++j) {
                    float* rp0 = scrw + (size_t)(m + j)      * 32 + rloc;
                    float* rp1 = scrw + (size_t)(m + j + 16) * 32 + rloc;
                    rp0[0]  = (*pA0)[j] * 0.03125f + sv0;  // 1/sqrt(1024)
                    rp1[0]  = (*pB0)[j] * 0.03125f + sv0;
                    rp0[16] = (*pA1)[j] * 0.03125f + sv1;
                    rp1[16] = (*pB1)[j] * 0.03125f + sv1;
                }
                asm volatile("s_wait_dscnt 0" ::: "memory");

                // every lane scans one of the 32 rows (32 cols)
                const float* rowp = scrw + lane * 32;
                const int idxBase = idx0 + sb + ch * 32;
                #pragma unroll 8
                for (int n = 0; n < 32; ++n) {
                    float v = rowp[n];
                    int idx = idxBase + n;
                    if (v > ts0) { ts3=ts2; ti3=ti2; ts2=ts1; ti2=ti1;
                                   ts1=ts0; ti1=ti0; ts0=v;  ti0=idx; }
                    else if (v > ts1) { ts3=ts2; ti3=ti2; ts2=ts1; ti2=ti1;
                                        ts1=v; ti1=idx; }
                    else if (v > ts2) { ts3=ts2; ti3=ti2; ts2=v; ti2=idx; }
                    else if (v > ts3) { ts3=v; ti3=idx; }
                }
                // same-wave LDS ops are in order: next chunk stores can't pass reads
            }
        }

        {   // per-(row, slot-half) candidate list -> workspace
            int row = rowBase + waveRow + lane;
            size_t base = (((size_t)level * NROWS + row) * 2 + wgrp) * 4;
            float* sp = topS + base;
            int*   ip = topI + base;
            sp[0]=ts0; sp[1]=ts1; sp[2]=ts2; sp[3]=ts3;
            ip[0]=ti0; ip[1]=ti1; ip[2]=ti2; ip[3]=ti3;
        }
    }
}

// ---------------- pass 3: merge candidates, softmax(top4) @ gathered V ------
__global__ __launch_bounds__(256)
void hmb_output(const float* __restrict__ V0,
                const float* __restrict__ V1,
                const float* __restrict__ V2,
                const float* __restrict__ topS,
                const int*   __restrict__ topI,
                float* __restrict__ out)
{
    const int row = blockIdx.x;
    const int d4  = threadIdx.x;            // 256 threads * float4 = 1024
    float4 acc = {0.f, 0.f, 0.f, 0.f};

    #pragma unroll
    for (int level = 0; level < 3; ++level) {
        const float* V  = (level == 0) ? V0 : (level == 1) ? V1 : V2;
        size_t base = ((size_t)level * NROWS + row) * 8;   // 2 groups x 4
        const float* sp = topS + base;                     // uniform -> s_load
        const int*   ip = topI + base;

        // merge the two disjoint candidate lists: top-4 of 8
        float b0 = -3.4e38f, b1 = -3.4e38f, b2 = -3.4e38f, b3 = -3.4e38f;
        int   j0 = 0, j1 = 0, j2 = 0, j3 = 0;
        #pragma unroll
        for (int c = 0; c < 8; ++c) {
            float v = sp[c]; int ix = ip[c];
            if (v > b0) { b3=b2;j3=j2; b2=b1;j2=j1; b1=b0;j1=j0; b0=v;j0=ix; }
            else if (v > b1) { b3=b2;j3=j2; b2=b1;j2=j1; b1=v;j1=ix; }
            else if (v > b2) { b3=b2;j3=j2; b2=v;j2=ix; }
            else if (v > b3) { b3=v;j3=ix; }
        }

        float m  = b0;                       // lists sorted desc -> b0 is max
        float w0 = __expf(b0 - m), w1 = __expf(b1 - m);
        float w2 = __expf(b2 - m), w3 = __expf(b3 - m);
        float inv = (1.0f / 3.0f) / (w0 + w1 + w2 + w3);
        w0 *= inv; w1 *= inv; w2 *= inv; w3 *= inv;

        float4 v0 = ((const float4*)(V + (size_t)j0 * D_MODEL))[d4];
        float4 v1 = ((const float4*)(V + (size_t)j1 * D_MODEL))[d4];
        float4 v2 = ((const float4*)(V + (size_t)j2 * D_MODEL))[d4];
        float4 v3 = ((const float4*)(V + (size_t)j3 * D_MODEL))[d4];
        acc.x += w0*v0.x + w1*v1.x + w2*v2.x + w3*v3.x;
        acc.y += w0*v0.y + w1*v1.y + w2*v2.y + w3*v3.y;
        acc.z += w0*v0.z + w1*v1.z + w2*v2.z + w3*v3.z;
        acc.w += w0*v0.w + w1*v1.w + w2*v2.w + w3*v3.w;
    }
    ((float4*)(out + (size_t)row * D_MODEL))[d4] = acc;
}

// ---------------------------------------------------------------------------
extern "C" void kernel_launch(void* const* d_in, const int* in_sizes, int n_in,
                              void* d_out, int out_size, void* d_ws, size_t ws_size,
                              hipStream_t stream) {
    const float* q  = (const float*)d_in[0];
    const float* K0 = (const float*)d_in[1];
    const float* V0 = (const float*)d_in[2];
    const float* s0 = (const float*)d_in[3];
    const float* K1 = (const float*)d_in[4];
    const float* V1 = (const float*)d_in[5];
    const float* s1 = (const float*)d_in[6];
    const float* K2 = (const float*)d_in[7];
    const float* V2 = (const float*)d_in[8];
    const float* s2 = (const float*)d_in[9];
    (void)in_sizes; (void)n_in; (void)out_size; (void)ws_size;

    // workspace layout
    char*   ws   = (char*)d_ws;
    __bf16* Kb   = (__bf16*)ws;                                   // 7,340,032 B
    float*  topS = (float*)(ws + (size_t)KTOT * D_MODEL * 2);     // 3,145,728 B
    int*    topI = (int*)((char*)topS + (size_t)3 * NROWS * 8 * 4);
    float*  out  = (float*)d_out;

    // pass 1: K banks -> bf16 (concatenated [3584][1024])
    hmb_cvt_bf16<<<(2048 * D_MODEL / 4 + 255) / 256, 256, 0, stream>>>(
        K0, Kb, 2048 * D_MODEL / 4);
    hmb_cvt_bf16<<<(1024 * D_MODEL / 4 + 255) / 256, 256, 0, stream>>>(
        K1, Kb + (size_t)2048 * D_MODEL, 1024 * D_MODEL / 4);
    hmb_cvt_bf16<<<(512 * D_MODEL / 4 + 255) / 256, 256, 0, stream>>>(
        K2, Kb + (size_t)3072 * D_MODEL, 512 * D_MODEL / 4);

    // pass 2: WMMA score GEMM + per-half top-4 (256 row-blocks of 128 rows)
    hmb_scores_topk<<<NROWS / ROWS_PER_WG, 256, SMEM_BYTES, stream>>>(
        q, Kb, s0, s1, s2, topS, topI);

    // pass 3: merge + softmax + V gather + blend
    hmb_output<<<NROWS, 256, 0, stream>>>(V0, V1, V2, topS, topI, out);
}
